// DwsConvBlock_23218593202765
// MI455X (gfx1250) — 8-bit matrix operand order repaired and checked by reading, compiled, not run on hardware
//
#include <hip/hip_runtime.h>
#include <hip/hip_bf16.h>
#include <stdint.h>

typedef __attribute__((ext_vector_type(8))) int v8i;
typedef __attribute__((ext_vector_type(4))) int v4i;
typedef __attribute__((ext_vector_type(2))) int v2i;

#define C_IN   256
#define C_OUT  512
#define HW     56
#define NTILES 49
#define ACT_PITCH 272

#define OFF_WDWQ   0
#define OFF_DWINV  16384
#define OFF_DWBIAS 17408
#define OFF_ALPHA  18432
#define OFF_PBIAS  20480
#define OFF_PW8    24576

__global__ void prep_dw(const float* __restrict__ w, const float* __restrict__ g,
                        const float* __restrict__ be, const float* __restrict__ mn,
                        const float* __restrict__ vr,
                        float* __restrict__ wdwq, float* __restrict__ dwinv,
                        float* __restrict__ dwbias) {
    int ci = blockIdx.x * blockDim.x + threadIdx.x;
    if (ci >= C_IN) return;
    const float* wc = w + ci * 9;
    float m = 0.f;
    #pragma unroll
    for (int j = 0; j < 9; ++j) m = fmaxf(m, fabsf(wc[j]));
    float s = fmaxf(m, 2e-16f) / 7.f;
    #pragma unroll
    for (int j = 0; j < 9; ++j) {
        float c = fminf(fmaxf(rintf(wc[j] / s), -7.f), 7.f);
        wdwq[ci * 16 + j] = c * s;
    }
    float inv = g[ci] * rsqrtf(vr[ci] + 1e-5f);
    dwinv[ci]  = inv;
    dwbias[ci] = be[ci] - mn[ci] * inv;
}

__global__ void prep_pw(const float* __restrict__ w, const float* __restrict__ g,
                        const float* __restrict__ be, const float* __restrict__ mn,
                        const float* __restrict__ vr,
                        signed char* __restrict__ pw8, float* __restrict__ alpha,
                        float* __restrict__ pbias) {
    int co = blockIdx.x * blockDim.x + threadIdx.x;
    if (co >= C_OUT) return;
    const float* wc = w + co * C_IN;
    float m = 0.f;
    for (int ci = 0; ci < C_IN; ++ci) m = fmaxf(m, fabsf(wc[ci]));
    float s = fmaxf(m, 2e-16f) / 7.f;
    for (int ci = 0; ci < C_IN; ++ci) {
        float c = fminf(fmaxf(rintf(wc[ci] / s), -7.f), 7.f);
        pw8[co * C_IN + ci] = (signed char)(int)c;
    }
    float inv = g[co] * rsqrtf(vr[co] + 1e-5f);
    alpha[co] = s * 0.4f * inv;
    pbias[co] = be[co] - mn[co] * inv;
}

__global__ __launch_bounds__(256, 1)
void dws_fused(const float* __restrict__ x,
               const float* __restrict__ wdwq, const float* __restrict__ dwinv,
               const float* __restrict__ dwbias,
               const signed char* __restrict__ pw8,
               const float* __restrict__ alpha, const float* __restrict__ pbias,
               float* __restrict__ out) {
    __shared__ unsigned char act[64 * ACT_PITCH];
    __shared__ float alpha_s[C_OUT];
    __shared__ float pbias_s[C_OUT];

    const int t   = threadIdx.x;
    const int blk = blockIdx.x;
    const int b   = blk / NTILES;
    const int tid = blk % NTILES;
    const int h0  = (tid / 7) * 8;
    const int w0  = (tid % 7) * 8;

    alpha_s[t]       = alpha[t];
    alpha_s[t + 256] = alpha[t + 256];
    pbias_s[t]       = pbias[t];
    pbias_s[t + 256] = pbias[t + 256];

    const int px = t & 63;
    const int cg = t >> 6;
    const int ty = px >> 3, tx = px & 7;

    int  soff[9];
    bool valid[9];
    #pragma unroll
    for (int dy = 0; dy < 3; ++dy)
        #pragma unroll
        for (int dx = 0; dx < 3; ++dx) {
            int yy = h0 + ty + dy - 1, xx = w0 + tx + dx - 1;
            bool v = (yy >= 0) && (yy < HW) && (xx >= 0) && (xx < HW);
            valid[dy * 3 + dx] = v;
            soff[dy * 3 + dx]  = v ? (yy * HW + xx) : 0;
        }
    const long xbase = (long)b * C_IN * (HW * HW);
    for (int k = 0; k < 64; ++k) {
        const int ci = cg * 64 + k;
        const float* xc = x + xbase + (long)ci * (HW * HW);
        const float* wc = wdwq + ci * 16;
        float s = 0.f;
        #pragma unroll
        for (int j = 0; j < 9; ++j) {
            float xv = valid[j] ? xc[soff[j]] : 0.f;
            s = fmaf(xv, wc[j], s);
        }
        float hbn = fmaf(s, dwinv[ci], dwbias[ci]);
        hbn = fmaxf(hbn, 0.f);
        float q = fminf(rintf(hbn * 2.5f), 15.f);
        act[px * ACT_PITCH + ci] = (unsigned char)(int)q;
    }
    __syncthreads();

    const int wave = t >> 5;
    const int lane = t & 31;
    const int lo16 = lane & 15;
    const int hih  = lane >> 4;

    for (int mt = wave; mt < 32; mt += 8) {
        const int co_base = mt * 16;
        v8i afr[4];
        #pragma unroll
        for (int kc = 0; kc < 4; ++kc) {
            const signed char* ap = pw8 + (co_base + lo16) * C_IN + kc * 64 + hih * 8;
            v2i a0 = *(const v2i*)(ap +  0);
            v2i a1 = *(const v2i*)(ap + 16);
            v2i a2 = *(const v2i*)(ap + 32);
            v2i a3 = *(const v2i*)(ap + 48);
            v8i af = {a0.x, a0.y, a1.x, a1.y, a2.x, a2.y, a3.x, a3.y};
            afr[kc] = af;
        }
        #pragma unroll
        for (int nt = 0; nt < 4; ++nt) {
            const int px_base = nt * 16;
            v8i acc = {0, 0, 0, 0, 0, 0, 0, 0};
            #pragma unroll
            for (int kc = 0; kc < 4; ++kc) {
                const unsigned char* bp =
                    act + (px_base + lo16) * ACT_PITCH + kc * 64 + hih * 8;
                v2i b0 = *(const v2i*)(bp);
                v2i b1 = *(const v2i*)(bp + 16);
                v2i b2 = *(const v2i*)(bp + 32);
                v2i b3 = *(const v2i*)(bp + 48);
                v8i bfr = {b0.x, b0.y, b1.x, b1.y, b2.x, b2.y, b3.x, b3.y};
                acc = __builtin_amdgcn_wmma_i32_16x16x64_iu8(
                          true, afr[kc], false, bfr, acc, false, false);
            }
            #pragma unroll
            for (int r = 0; r < 8; ++r) {
                const int co = co_base + r + hih * 8;
                const int p  = px_base + lo16;
                float v = fmaf((float)acc[r], alpha_s[co], pbias_s[co]);
                v = fmaxf(v, 0.f);
                float q = fminf(rintf(v * 2.5f), 15.f);
                long oidx = (((long)b * C_OUT + co) * HW + (h0 + (p >> 3))) * HW
                            + (w0 + (p & 7));
                out[oidx] = q * 0.4f;
            }
        }
    }
}

extern "C" void kernel_launch(void* const* d_in, const int* in_sizes, int n_in,
                              void* d_out, int out_size, void* d_ws, size_t ws_size,
                              hipStream_t stream) {
    const float* x        = (const float*)d_in[0];
    const float* dw_w     = (const float*)d_in[1];
    const float* dw_gamma = (const float*)d_in[2];
    const float* dw_beta  = (const float*)d_in[3];
    const float* dw_mean  = (const float*)d_in[4];
    const float* dw_var   = (const float*)d_in[5];
    const float* pw_w     = (const float*)d_in[6];
    const float* pw_gamma = (const float*)d_in[7];
    const float* pw_beta  = (const float*)d_in[8];
    const float* pw_mean  = (const float*)d_in[9];
    const float* pw_var   = (const float*)d_in[10];
    float* out = (float*)d_out;

    char* ws = (char*)d_ws;
    float*       wdwq   = (float*)(ws + OFF_WDWQ);
    float*       dwinvp = (float*)(ws + OFF_DWINV);
    float*       dwbias = (float*)(ws + OFF_DWBIAS);
    float*       alphap = (float*)(ws + OFF_ALPHA);
    float*       pbiasp = (float*)(ws + OFF_PBIAS);
    signed char* pw8    = (signed char*)(ws + OFF_PW8);

    prep_dw<<<1, 256, 0, stream>>>(dw_w, dw_gamma, dw_beta, dw_mean, dw_var,
                                   wdwq, dwinvp, dwbias);
    prep_pw<<<2, 256, 0, stream>>>(pw_w, pw_gamma, pw_beta, pw_mean, pw_var,
                                   pw8, alphap, pbiasp);
    dws_fused<<<32 * NTILES, 256, 0, stream>>>(x, wdwq, dwinvp, dwbias,
                                               pw8, alphap, pbiasp, out);
}
